// Attention_30855045054988
// MI455X (gfx1250) — matrix operand order repaired and checked by reading, not compiled, not run on hardware
//
#include <hip/hip_runtime.h>
#include <hip/hip_bf16.h>
#include <cstdint>
#include <cstddef>

// Problem constants (match reference)
#define DIMC   1024
#define NHEAD  16
#define HDIM   64
#define SEQ    2048
#define BATCH  2
#define ROWS   (BATCH * SEQ)      // 4096 token rows
#define QKVOUT (3 * DIMC)         // 3072
#define QSCALE 0.125f             // HDIM^-0.5

typedef __attribute__((ext_vector_type(16))) __bf16 v16bf;
typedef __attribute__((ext_vector_type(8)))  float  v8f;

// ---------- helpers ----------

// RNE fp32 -> bf16 without relying on target cvt codegen
static __device__ __forceinline__ __bf16 to_bf16(float f) {
    union { float f; unsigned u; } a; a.f = f;
    unsigned r = a.u + 0x7FFFu + ((a.u >> 16) & 1u);
    unsigned short h = (unsigned short)(r >> 16);
    return __builtin_bit_cast(__bf16, h);
}

// Two 16-byte loads -> one 32-byte WMMA fragment (8 VGPRs)
static __device__ __forceinline__ v16bf load_frag_pair(const __bf16* p0, const __bf16* p1) {
    union { v16bf v; uint4 q[2]; } u;
    u.q[0] = *reinterpret_cast<const uint4*>(p0);
    u.q[1] = *reinterpret_cast<const uint4*>(p1);
    return u.v;
}

static __device__ __forceinline__ v8f wmma_bf16(v16bf a, v16bf b, v8f c) {
    // D(f32 16x16) = A(bf16 16x32) * B(bf16 32x16) + C
    return __builtin_amdgcn_wmma_f32_16x16x32_bf16(
        /*neg_a=*/false, a, /*neg_b=*/false, b,
        /*c_mod=*/(short)0, c, /*reuse_a=*/false, /*reuse_b=*/false);
}

// ---------- kernel 1: fp32 -> bf16 convert ----------

__global__ __launch_bounds__(256) void k_cvt_bf16(const float* __restrict__ in,
                                                  __bf16* __restrict__ out, int n) {
    int i = blockIdx.x * blockDim.x + threadIdx.x;
    int stride = gridDim.x * blockDim.x;
    for (; i < n; i += stride) out[i] = to_bf16(in[i]);
}

// ---------- software-pipelined 16x64 GEMM tile core ----------
// acc[t] += A(rows m0..m0+15) * B-cols (n0 + t*16 .. +15), K = DIMC.
// Explicit ping-pong between two fragment buffer sets, loop unrolled by
// double K-steps: no buffer-rotation register copies, and each K-step's
// loads are in flight while the previous step's 4 WMMAs issue.

static __device__ __forceinline__ void load_step(const __bf16* __restrict__ arow,
                                                 const __bf16* const brow[4],
                                                 int kb, v16bf& a, v16bf b[4]) {
    a = load_frag_pair(arow + kb, arow + kb + 16);
#pragma unroll
    for (int t = 0; t < 4; ++t)
        b[t] = load_frag_pair(brow[t] + kb, brow[t] + kb + 16);
}

static __device__ __forceinline__ void gemm_tile_16x64(const __bf16* __restrict__ arow,
                                                       const __bf16* const brow[4],
                                                       v8f acc[4]) {
    v16bf aP, bP[4];        // ping buffers
    v16bf aQ, bQ[4];        // pong buffers

    load_step(arow, brow, 0, aP, bP);                        // K-step 0

    // 32 K-steps total; handle steps [0,30) in double-steps, peel the tail.
    for (int kb = 0; kb < DIMC - 64; kb += 64) {
        __builtin_prefetch(arow + kb + 128, 0, 1);           // global_prefetch_b8
        load_step(arow, brow, kb + 32, aQ, bQ);              // next step -> pong
#pragma unroll
        for (int t = 0; t < 4; ++t)
            acc[t] = wmma_bf16(aP, bP[t], acc[t]);           // consume ping
        load_step(arow, brow, kb + 64, aP, bP);              // step after -> ping
#pragma unroll
        for (int t = 0; t < 4; ++t)
            acc[t] = wmma_bf16(aQ, bQ[t], acc[t]);           // consume pong
    }
    // tail: steps 30 and 31 (ping holds step 30 from the last loop iteration)
    load_step(arow, brow, DIMC - 32, aQ, bQ);
#pragma unroll
    for (int t = 0; t < 4; ++t)
        acc[t] = wmma_bf16(aP, bP[t], acc[t]);
#pragma unroll
    for (int t = 0; t < 4; ++t)
        acc[t] = wmma_bf16(aQ, bQ[t], acc[t]);
}

// ---------- kernel 2: QKV projection  qkv[r, d] = X[r,:] . Wqkv[d,:] ----------
// Each wave: 16 rows x 64 cols, K=1024 in 32-wide steps (4 WMMAs / step).
// Scatter into Q[b,h,n,dh]*scale, K[b,h,n,dh], VT[b,h,dh,n] (bf16).

__global__ __launch_bounds__(256) void k_gemm_qkv(const __bf16* __restrict__ X,
                                                  const __bf16* __restrict__ W,
                                                  __bf16* __restrict__ Q,
                                                  __bf16* __restrict__ K,
                                                  __bf16* __restrict__ VT) {
    const int lane  = threadIdx.x & 31;
    const int wid   = threadIdx.x >> 5;
    const int col16 = lane & 15;
    const int half  = lane >> 4;

    const int NT   = QKVOUT / 64;                 // 48 column groups
    const int wave = blockIdx.x * 8 + wid;        // 256*48 = 12288 waves total
    const int m0   = (wave / NT) * 16;
    const int n0   = (wave % NT) * 64;

    v8f acc[4] = { {}, {}, {}, {} };

    const __bf16* arow = X + (size_t)(m0 + col16) * DIMC + half * 8;  // A frag base
    const __bf16* brow[4];
#pragma unroll
    for (int t = 0; t < 4; ++t)
        brow[t] = W + (size_t)(n0 + t * 16 + col16) * DIMC + half * 8;

    gemm_tile_16x64(arow, brow, acc);

    const int which = n0 / DIMC;                 // 0=q 1=k 2=v (uniform per wave)
    const int h     = (n0 % DIMC) / HDIM;        // uniform per wave
#pragma unroll
    for (int t = 0; t < 4; ++t) {
        int dh = t * 16 + col16;
#pragma unroll
        for (int r = 0; r < 8; ++r) {
            int row = m0 + r + half * 8;         // D layout: M = r + half*8
            int b   = row >> 11;
            int tok = row & (SEQ - 1);
            float v = acc[t][r];
            if (which == 0) {
                Q[((size_t)(b * NHEAD + h) * SEQ + tok) * HDIM + dh] = to_bf16(v * QSCALE);
            } else if (which == 1) {
                K[((size_t)(b * NHEAD + h) * SEQ + tok) * HDIM + dh] = to_bf16(v);
            } else {
                VT[((size_t)(b * NHEAD + h) * HDIM + dh) * SEQ + tok] = to_bf16(v);
            }
        }
    }
}

// ---------- kernel 3: flash attention ----------
// One wave = one (b,h, 16-query tile). Stream keys 32 at a time:
//   S(16x32) = Qtile(16x64) @ K^T  -> 4 WMMAs
//   V fragments issued BEFORE softmax so their latency hides under the VALU
//   online softmax (shfl_xor row reductions within 16-lane halves)
//   P repacked D->A layout via per-wave LDS (1 KB), s_wait_dscnt, ds_load_b128
//   O(16x64) += P(16x32) @ V      -> 4 WMMAs
// Output attn_out[b, tok, h*64+dh] in bf16.

__global__ __launch_bounds__(256) void k_attn(const __bf16* __restrict__ Q,
                                              const __bf16* __restrict__ K,
                                              const __bf16* __restrict__ VT,
                                              __bf16* __restrict__ AO) {
    __shared__ __bf16 smem[8][16 * 32];          // 1 KB per wave

    const int lane  = threadIdx.x & 31;
    const int wid   = threadIdx.x >> 5;
    const int col16 = lane & 15;
    const int half  = lane >> 4;

    const int wave = blockIdx.x * 8 + wid;       // 2*16*128 = 4096 waves total
    const int qt   = wave & 127;
    const int bh   = wave >> 7;
    const int h    = bh & (NHEAD - 1);
    const int b    = bh >> 4;
    const int q0   = qt * 16;

    const __bf16* Qp  = Q  + (size_t)bh * SEQ * HDIM;
    const __bf16* Kp  = K  + (size_t)bh * SEQ * HDIM;
    const __bf16* VTp = VT + (size_t)bh * HDIM * SEQ;
    __bf16* sm = &smem[wid][0];

    // Q fragments (A layout), covering Dh = 0..31 and 32..63; scale pre-folded.
    const __bf16* qrow = Qp + (size_t)(q0 + col16) * HDIM + half * 8;
    const v16bf qf0 = load_frag_pair(qrow,      qrow + 16);
    const v16bf qf1 = load_frag_pair(qrow + 32, qrow + 48);

    float mrow[8], lrow[8];
#pragma unroll
    for (int r = 0; r < 8; ++r) { mrow[r] = -__builtin_inff(); lrow[r] = 0.0f; }
    v8f oacc[4] = { {}, {}, {}, {} };

    for (int j0 = 0; j0 < SEQ; j0 += 32) {
        // ---- scores: two 16x16 tiles over keys [j0, j0+32) ----
        const __bf16* kb0 = Kp + (size_t)(j0      + col16) * HDIM + half * 8;
        const __bf16* kb1 = Kp + (size_t)(j0 + 16 + col16) * HDIM + half * 8;
        v8f s0 = {}, s1 = {};
        s0 = wmma_bf16(qf0, load_frag_pair(kb0,      kb0 + 16), s0);
        s0 = wmma_bf16(qf1, load_frag_pair(kb0 + 32, kb0 + 48), s0);
        s1 = wmma_bf16(qf0, load_frag_pair(kb1,      kb1 + 16), s1);
        s1 = wmma_bf16(qf1, load_frag_pair(kb1 + 32, kb1 + 48), s1);

        // ---- issue V fragments now; consumed only after the LDS round-trip,
        // so their global latency hides under the softmax VALU below ----
        v16bf vfrag[4];
#pragma unroll
        for (int t = 0; t < 4; ++t) {
            const __bf16* vb = VTp + (size_t)(t * 16 + col16) * SEQ + half * 8 + j0;
            vfrag[t] = load_frag_pair(vb, vb + 16);
        }

        // ---- online softmax (row = r + half*8, columns across 16 lanes) ----
#pragma unroll
        for (int r = 0; r < 8; ++r) {
            float a = s0[r], c = s1[r];
            float mx = fmaxf(a, c);
            mx = fmaxf(mx, __shfl_xor(mx, 1));
            mx = fmaxf(mx, __shfl_xor(mx, 2));
            mx = fmaxf(mx, __shfl_xor(mx, 4));
            mx = fmaxf(mx, __shfl_xor(mx, 8));
            float mnew  = fmaxf(mrow[r], mx);
            float alpha = __expf(mrow[r] - mnew);
            float p0 = __expf(a - mnew);
            float p1 = __expf(c - mnew);
            float s  = p0 + p1;
            s += __shfl_xor(s, 1);
            s += __shfl_xor(s, 2);
            s += __shfl_xor(s, 4);
            s += __shfl_xor(s, 8);
            lrow[r] = lrow[r] * alpha + s;
            mrow[r] = mnew;
            oacc[0][r] *= alpha; oacc[1][r] *= alpha;
            oacc[2][r] *= alpha; oacc[3][r] *= alpha;
            // stage P (D layout -> row-major 16x32 in LDS)
            int row = r + half * 8;
            sm[row * 32 + col16]      = to_bf16(p0);
            sm[row * 32 + 16 + col16] = to_bf16(p1);
        }

        // all 32 lanes' ds_store_b16 must land before cross-lane ds_load_b128
        asm volatile("s_wait_dscnt 0" ::: "memory");

        const __bf16* prow = sm + col16 * 32 + half * 8;
        v16bf pf = load_frag_pair(prow, prow + 16);   // A layout, K = 32 keys

        // ---- O += P @ V : 4 Dh tiles, B frags contiguous thanks to VT ----
#pragma unroll
        for (int t = 0; t < 4; ++t)
            oacc[t] = wmma_bf16(pf, vfrag[t], oacc[t]);
    }

    // ---- normalize and emit attn_out[b, tok, h*64 + dh] ----
#pragma unroll
    for (int t = 0; t < 4; ++t) {
        int col = h * HDIM + t * 16 + col16;
#pragma unroll
        for (int r = 0; r < 8; ++r) {
            int tok = q0 + r + half * 8;
            AO[((size_t)b * SEQ + tok) * DIMC + col] = to_bf16(oacc[t][r] / lrow[r]);
        }
    }
}

// ---------- kernel 4: output projection  out = AO @ Wproj^T + b ----------

__global__ __launch_bounds__(256) void k_gemm_proj(const __bf16* __restrict__ A,
                                                   const __bf16* __restrict__ W,
                                                   const float* __restrict__ bias,
                                                   float* __restrict__ out) {
    const int lane  = threadIdx.x & 31;
    const int wid   = threadIdx.x >> 5;
    const int col16 = lane & 15;
    const int half  = lane >> 4;

    const int NT   = DIMC / 64;                   // 16 column groups
    const int wave = blockIdx.x * 8 + wid;        // 256*16 = 4096 waves total
    const int m0   = (wave / NT) * 16;
    const int n0   = (wave % NT) * 64;

    v8f acc[4] = { {}, {}, {}, {} };

    const __bf16* arow = A + (size_t)(m0 + col16) * DIMC + half * 8;
    const __bf16* brow[4];
#pragma unroll
    for (int t = 0; t < 4; ++t)
        brow[t] = W + (size_t)(n0 + t * 16 + col16) * DIMC + half * 8;

    gemm_tile_16x64(arow, brow, acc);

#pragma unroll
    for (int t = 0; t < 4; ++t) {
        int col = n0 + t * 16 + col16;
        float bv = bias[col];
#pragma unroll
        for (int r = 0; r < 8; ++r) {
            int row = m0 + r + half * 8;
            out[(size_t)row * DIMC + col] = acc[t][r] + bv;
        }
    }
}

// ---------- host side ----------

extern "C" void kernel_launch(void* const* d_in, const int* in_sizes, int n_in,
                              void* d_out, int out_size, void* d_ws, size_t ws_size,
                              hipStream_t stream) {
    (void)in_sizes; (void)n_in; (void)out_size; (void)ws_size;

    const float* x     = (const float*)d_in[0];   // [2,2048,1024]
    const float* wqkv  = (const float*)d_in[1];   // [3072,1024]
    const float* wproj = (const float*)d_in[2];   // [1024,1024]
    const float* bproj = (const float*)d_in[3];   // [1024]
    float* out = (float*)d_out;                   // [2,2048,1024] fp32

    // bf16 staging carved from d_ws (48 MB total)
    __bf16* xb  = (__bf16*)d_ws;
    __bf16* wqb = xb  + (size_t)ROWS * DIMC;                 //  8 MB
    __bf16* wpb = wqb + (size_t)QKVOUT * DIMC;               //  6 MB
    __bf16* qb  = wpb + (size_t)DIMC * DIMC;                 //  2 MB
    __bf16* kb  = qb  + (size_t)BATCH * NHEAD * SEQ * HDIM;  //  8 MB
    __bf16* vtb = kb  + (size_t)BATCH * NHEAD * SEQ * HDIM;  //  8 MB
    __bf16* aob = vtb + (size_t)BATCH * NHEAD * SEQ * HDIM;  //  8 MB (+8 MB)

    // 1) fp32 -> bf16 converts
    k_cvt_bf16<<<(ROWS * DIMC + 255) / 256, 256, 0, stream>>>(x, xb, ROWS * DIMC);
    k_cvt_bf16<<<(QKVOUT * DIMC + 255) / 256, 256, 0, stream>>>(wqkv, wqb, QKVOUT * DIMC);
    k_cvt_bf16<<<(DIMC * DIMC + 255) / 256, 256, 0, stream>>>(wproj, wpb, DIMC * DIMC);

    // 2) QKV projection: 256 row-tiles x 48 col-groups, 8 waves/block
    k_gemm_qkv<<<(256 * 48) / 8, 256, 0, stream>>>(xb, wqb, qb, kb, vtb);

    // 3) flash attention: 2*16*128 waves, 8 waves/block
    k_attn<<<(BATCH * NHEAD * (SEQ / 16)) / 8, 256, 0, stream>>>(qb, kb, vtb, aob);

    // 4) output projection + bias: 256 x 16 wave tiles, 8 waves/block
    k_gemm_proj<<<(256 * 16) / 8, 256, 0, stream>>>(aob, wpb, bproj, out);
}
